// MMRecBlock_77876347011649
// MI455X (gfx1250) — matrix operand order repaired and checked by reading, not compiled, not run on hardware
//
#include <hip/hip_runtime.h>

// ---------------------------------------------------------------------------
// CDNA5 (gfx1250) MM-Rec block. wave32, WMMA bf16 (f32 accum) for all GEMMs.
// GEMM B-tiles staged through LDS with async copies (ASYNCcnt), double-buffered.
// Attention loads batched/double-buffered to overlap with WMMA.
// ---------------------------------------------------------------------------

typedef __bf16 bf16;
typedef __attribute__((ext_vector_type(16))) bf16 bf16x16;
typedef __attribute__((ext_vector_type(8)))  float f32x8;
typedef __attribute__((ext_vector_type(4)))  unsigned int u32x4;

union FragBF {            // 16 bf16 = 32 bytes = two b128 loads
  u32x4  u[2];
  bf16x16 v;
};

__device__ __forceinline__ float sigmoidf_(float x) { return 1.f / (1.f + __expf(-x)); }

// ---------------------------------------------------------------------------
// Weight convert+transpose: W fp32 [K][N] -> Wt bf16 [N][K]
// ---------------------------------------------------------------------------
__global__ void transpose_w_kernel(const float* __restrict__ W, bf16* __restrict__ Wt,
                                   int K, int N) {
  size_t idx = (size_t)blockIdx.x * blockDim.x + threadIdx.x;
  if (idx >= (size_t)K * N) return;
  int k = (int)(idx / N);
  int n = (int)(idx % N);
  Wt[(size_t)n * K + k] = (bf16)W[idx];
}

// ---------------------------------------------------------------------------
// RMSNorm: out_bf16 = w * x / (||x||/sqrt(D) + eps). One block per row.
// ---------------------------------------------------------------------------
__global__ __launch_bounds__(256)
void rmsnorm_kernel(const float* __restrict__ x, const float* __restrict__ w,
                    bf16* __restrict__ out, int D) {
  const int row = blockIdx.x;
  const float* xr = x + (size_t)row * D;
  __shared__ float red[8];
  __shared__ float scl;
  float ss = 0.f;
  for (int i = threadIdx.x; i < D; i += 256) { float v = xr[i]; ss += v * v; }
#pragma unroll
  for (int m = 16; m >= 1; m >>= 1) ss += __shfl_xor(ss, m, 32);
  if ((threadIdx.x & 31) == 0) red[threadIdx.x >> 5] = ss;
  __syncthreads();
  if (threadIdx.x == 0) {
    float t = 0.f;
#pragma unroll
    for (int i = 0; i < 8; ++i) t += red[i];
    float norm = sqrtf(t) / sqrtf((float)D);
    scl = 1.f / (norm + 1e-6f);
  }
  __syncthreads();
  const float s = scl;
  for (int i = threadIdx.x; i < D; i += 256)
    out[(size_t)row * D + i] = (bf16)(w[i] * xr[i] * s);
}

// ---------------------------------------------------------------------------
// Generic WMMA GEMM: out = epi(A[M,K] @ Wt[N,K]^T + bias), fp32 and/or bf16 out.
// Block = 4 waves; wave computes 16 rows x 64 cols; grid = (N/64, M/64).
// B tile (64 cols x 32 K) staged via global_load_async_to_lds_b128, double-
// buffered; A fragments software-pipelined in registers.
// ---------------------------------------------------------------------------
enum { EPI_NONE = 0, EPI_RELU = 1, EPI_GELU = 2, EPI_SIGCLIP = 3, EPI_RESID = 4 };

template <int EPI>
__global__ __launch_bounds__(128)
void gemm_wmma_kernel(const bf16* __restrict__ A, const bf16* __restrict__ Wt,
                      const float* __restrict__ bias, const float* __restrict__ resid,
                      float* __restrict__ outF, bf16* __restrict__ outB,
                      int M, int N, int K) {
  const int tid  = threadIdx.x;
  const int lane = tid & 31;
  const int wave = tid >> 5;
  const int l15  = lane & 15;
  const int h    = lane >> 4;
  const int mblk = blockIdx.y * 64 + wave * 16;
  const int nblk = blockIdx.x * 64;

  // [buf][col * 32 + k] bf16 : 64B per column -> each lane's B fragment is
  // two aligned b128 LDS reads.
  __shared__ bf16 Bsm[2][64 * 32];

  // Async-copy one 64x32 B tile (4KB = 256 x 16B chunks, 2 chunks/thread).
  auto issue_tile = [&](int kk, int buf) {
#pragma unroll
    for (int t = 0; t < 2; ++t) {
      const int ci  = tid + t * 128;
      const int col = ci >> 2;          // 4 chunks per column
      const int sub = ci & 3;
      const bf16* gp = Wt + (size_t)(nblk + col) * K + kk + sub * 8;
      const unsigned lds_off = (unsigned)(size_t)(&Bsm[buf][col * 32 + sub * 8]);
      asm volatile("global_load_async_to_lds_b128 %0, %1, off"
                   :: "v"(lds_off), "v"((unsigned long long)(size_t)gp)
                   : "memory");
    }
  };

  const f32x8 z8 = {0.f, 0.f, 0.f, 0.f, 0.f, 0.f, 0.f, 0.f};
  f32x8 acc[4];
#pragma unroll
  for (int j = 0; j < 4; ++j) acc[j] = z8;

  const bf16* arow = A + (size_t)(mblk + l15) * K;

  // Prologue: tile 0 in flight, A fragment 0 in registers.
  issue_tile(0, 0);
  FragBF a;
  a.u[0] = *reinterpret_cast<const u32x4*>(arow + h * 8);
  a.u[1] = *reinterpret_cast<const u32x4*>(arow + 16 + h * 8);

  const int nk = K / 32;
  for (int ik = 0; ik < nk; ++ik) {
    const int kk  = ik * 32;
    const int buf = ik & 1;
    if (ik + 1 < nk) {
      issue_tile(kk + 32, buf ^ 1);                       // next tile in flight
      asm volatile("s_wait_asynccnt 0x2" ::: "memory");   // current tile landed
    } else {
      asm volatile("s_wait_asynccnt 0x0" ::: "memory");
    }
    __syncthreads();

    FragBF anext;
    if (ik + 1 < nk) {                                    // pipeline A loads
      anext.u[0] = *reinterpret_cast<const u32x4*>(arow + kk + 32 + h * 8);
      anext.u[1] = *reinterpret_cast<const u32x4*>(arow + kk + 48 + h * 8);
      __builtin_prefetch(arow + kk + 64 + h * 8, 0, 1);
    }

#pragma unroll
    for (int j = 0; j < 4; ++j) {
      const bf16* bp = &Bsm[buf][(j * 16 + l15) * 32 + h * 8];
      FragBF b;
      b.u[0] = *reinterpret_cast<const u32x4*>(bp);
      b.u[1] = *reinterpret_cast<const u32x4*>(bp + 16);
      acc[j] = __builtin_amdgcn_wmma_f32_16x16x32_bf16(false, a.v, false, b.v,
                                                       (short)0, acc[j], false, false);
    }
    if (ik + 1 < nk) a = anext;
    __syncthreads();   // don't let next tile's async writes race readers
  }

#pragma unroll
  for (int j = 0; j < 4; ++j) {
    const int col = nblk + j * 16 + l15;
    const float bi = bias ? bias[col] : 0.f;
#pragma unroll
    for (int v = 0; v < 8; ++v) {
      const int row = mblk + v + 8 * h;   // C/D layout: m = v + 8*(lane>>4)
      float val = acc[j][v] + bi;
      if (EPI == EPI_RELU) {
        val = fmaxf(val, 0.f);
      } else if (EPI == EPI_GELU) {
        const float c = 0.7978845608f * (val + 0.044715f * val * val * val);
        val = 0.5f * val * (1.f + tanhf(c));
      } else if (EPI == EPI_SIGCLIP) {
        val = sigmoidf_(val);
        val = fminf(fmaxf(val, 1e-6f), 1.f - 1e-6f);
      } else if (EPI == EPI_RESID) {
        val += resid[(size_t)row * N + col];
      }
      if (outF) outF[(size_t)row * N + col] = val;
      if (outB) outB[(size_t)row * N + col] = (bf16)val;
    }
  }
}

// ---------------------------------------------------------------------------
// v bf16 [B,S,D] -> vt bf16 [B,H,dh,S] (time-contiguous for PV B-fragments)
// ---------------------------------------------------------------------------
__global__ void transpose_v_kernel(const bf16* __restrict__ v, bf16* __restrict__ vt,
                                   int B, int S, int D, int dh) {
  size_t idx = (size_t)blockIdx.x * blockDim.x + threadIdx.x;
  if (idx >= (size_t)B * S * D) return;
  int d = (int)(idx % D);
  int s = (int)((idx / D) % S);
  int b = (int)(idx / ((size_t)D * S));
  int hh = d / dh, dd = d % dh;
  int H = D / dh;
  vt[(((size_t)b * H + hh) * dh + dd) * S + s] = v[idx];
}

// ---------------------------------------------------------------------------
// a_t = exp(cumsum(log gamma)); ks = bf16(k * a). Block per batch, thread per d.
// ---------------------------------------------------------------------------
__global__ __launch_bounds__(1024)
void scan_scale_kernel(const float* __restrict__ gamma, const float* __restrict__ kf,
                       bf16* __restrict__ ks, int S, int D) {
  const int b = blockIdx.x;
  const int d = threadIdx.x;
  float lg = 0.f;
  for (int s = 0; s < S; ++s) {
    const size_t idx = (size_t)(b * S + s) * D + d;
    lg += __logf(gamma[idx]);
    ks[idx] = (bf16)(kf[idx] * __expf(lg));
  }
}

// ---------------------------------------------------------------------------
// Gated recurrence: h_t = z_t * sigmoid(h_{t-1} @ Wg + bg) + gamma_t * h_{t-1}
// One block per batch (1024 threads), h_{t-1} broadcast through LDS.
// ---------------------------------------------------------------------------
__global__ __launch_bounds__(1024)
void recurrence_kernel(const float* __restrict__ z, const float* __restrict__ gamma,
                       const bf16* __restrict__ Wgt, const float* __restrict__ bg,
                       float* __restrict__ hseq, int S, int D) {
  const int b = blockIdx.x;
  const int d = threadIdx.x;
  __shared__ float hprev[1024];
  hprev[d] = 0.f;
  __syncthreads();
  const bf16* wrow = Wgt + (size_t)d * D;   // Wgt[N=d][K] = Wg[k][d]
  const float bgd = bg[d];
  for (int t = 0; t < S; ++t) {
    float dot = bgd;
    for (int k = 0; k < D; k += 4) {
      dot += hprev[k    ] * (float)wrow[k    ]
           + hprev[k + 1] * (float)wrow[k + 1]
           + hprev[k + 2] * (float)wrow[k + 2]
           + hprev[k + 3] * (float)wrow[k + 3];
    }
    const size_t idx = (size_t)(b * S + t) * D + d;
    const float hn = z[idx] * sigmoidf_(dot) + gamma[idx] * hprev[d];
    __syncthreads();
    hprev[d] = hn;
    hseq[idx] = hn;
    __syncthreads();
  }
}

// ---------------------------------------------------------------------------
// Flash attention (causal). One wave per (b, head, 16-query block).
// QK^T and PV both via v_wmma_f32_16x16x32_bf16; P transposed through LDS.
// K fragments batched per tile; V fragments double-buffered across WMMAs.
// ---------------------------------------------------------------------------
__global__ __launch_bounds__(32)
void attn_kernel(const bf16* __restrict__ Q, const bf16* __restrict__ Ks,
                 const bf16* __restrict__ Vt, float* __restrict__ ctx,
                 int B, int H, int S, int dh) {
  const int q0   = blockIdx.x * 16;
  const int head = blockIdx.y;
  const int b    = blockIdx.z;
  const int lane = threadIdx.x & 31;
  const int l15  = lane & 15;
  const int h    = lane >> 4;
  const int D    = H * dh;
  const float scale = rsqrtf((float)dh);

  __shared__ bf16 Pl[16 * 32];

  // Q fragments (A-layout) for the 4 K-chunks of dh=128; reused every block.
  FragBF qa[4];
  const bf16* qrow = Q + ((size_t)(b * S + q0 + l15) * D + head * dh);
#pragma unroll
  for (int kc = 0; kc < 4; ++kc) {
    qa[kc].u[0] = *reinterpret_cast<const u32x4*>(qrow + kc * 32 + h * 8);
    qa[kc].u[1] = *reinterpret_cast<const u32x4*>(qrow + kc * 32 + 16 + h * 8);
  }

  // Loop-invariant base pointers (per-tile address = base + scaled offset).
  const bf16* kbase = Ks + ((size_t)(b * S + l15) * D + head * dh + h * 8);
  const bf16* vbase = Vt + ((size_t)(b * H + head) * dh * S + (size_t)l15 * S + h * 8);

  const f32x8 z8 = {0.f, 0.f, 0.f, 0.f, 0.f, 0.f, 0.f, 0.f};
  float Mrow[8], Lrow[8];
  f32x8 O[8];
#pragma unroll
  for (int v = 0; v < 8; ++v) { Mrow[v] = -3.0e38f; Lrow[v] = 0.f; }
#pragma unroll
  for (int j = 0; j < 8; ++j) O[j] = z8;

  for (int kb = 0; kb <= q0; kb += 32) {
    // ---- scores: two 16x16 key tiles; batch loads, then WMMAs ----
    f32x8 s[2];
    s[0] = z8; s[1] = z8;
#pragma unroll
    for (int t = 0; t < 2; ++t) {
      const bf16* krow = kbase + (size_t)(kb + 16 * t) * D;
      FragBF kf[4];
#pragma unroll
      for (int kc = 0; kc < 4; ++kc) {
        kf[kc].u[0] = *reinterpret_cast<const u32x4*>(krow + kc * 32);
        kf[kc].u[1] = *reinterpret_cast<const u32x4*>(krow + kc * 32 + 16);
      }
#pragma unroll
      for (int kc = 0; kc < 4; ++kc)
        s[t] = __builtin_amdgcn_wmma_f32_16x16x32_bf16(false, qa[kc].v, false, kf[kc].v,
                                                       (short)0, s[t], false, false);
    }
    // ---- online softmax (rows live in VGPR v for lane half h) ----
#pragma unroll
    for (int v = 0; v < 8; ++v) {
      const int qr = q0 + v + 8 * h;
#pragma unroll
      for (int t = 0; t < 2; ++t) {
        const int key = kb + t * 16 + l15;
        s[t][v] = (key <= qr) ? s[t][v] * scale : -3.0e38f;
      }
      float r = fmaxf(s[0][v], s[1][v]);
#pragma unroll
      for (int mm = 8; mm >= 1; mm >>= 1) r = fmaxf(r, __shfl_xor(r, mm, 16));
      const float newM = fmaxf(Mrow[v], r);
      const float corr = __expf(Mrow[v] - newM);
      const float p0 = __expf(s[0][v] - newM);
      const float p1 = __expf(s[1][v] - newM);
      float ls = p0 + p1;
#pragma unroll
      for (int mm = 8; mm >= 1; mm >>= 1) ls += __shfl_xor(ls, mm, 16);
      Lrow[v] = Lrow[v] * corr + ls;
      Mrow[v] = newM;
#pragma unroll
      for (int j = 0; j < 8; ++j) O[j][v] *= corr;
      Pl[(v + 8 * h) * 32 + l15]      = (bf16)p0;
      Pl[(v + 8 * h) * 32 + 16 + l15] = (bf16)p1;
    }
    __syncthreads();
    // ---- PV: reload P in A-fragment layout from LDS; V frags double-buffered
    FragBF pa;
    pa.u[0] = *reinterpret_cast<const u32x4*>(&Pl[l15 * 32 + h * 8]);
    pa.u[1] = *reinterpret_cast<const u32x4*>(&Pl[l15 * 32 + 16 + h * 8]);

    FragBF vf[2];
    const bf16* vrow0 = vbase + kb;
    vf[0].u[0] = *reinterpret_cast<const u32x4*>(vrow0);
    vf[0].u[1] = *reinterpret_cast<const u32x4*>(vrow0 + 16);
#pragma unroll
    for (int j = 0; j < 8; ++j) {
      if (j + 1 < 8) {
        const bf16* vrow = vbase + (size_t)(j + 1) * 16 * S + kb;
        vf[(j + 1) & 1].u[0] = *reinterpret_cast<const u32x4*>(vrow);
        vf[(j + 1) & 1].u[1] = *reinterpret_cast<const u32x4*>(vrow + 16);
      }
      O[j] = __builtin_amdgcn_wmma_f32_16x16x32_bf16(false, pa.v, false, vf[j & 1].v,
                                                     (short)0, O[j], false, false);
    }
    __syncthreads();
  }
  // ---- normalize + store ----
#pragma unroll
  for (int v = 0; v < 8; ++v) {
    const float inv = 1.f / Lrow[v];
    const int row = q0 + v + 8 * h;
#pragma unroll
    for (int j = 0; j < 8; ++j)
      ctx[(size_t)(b * S + row) * D + head * dh + j * 16 + l15] = O[j][v] * inv;
  }
}

// ---------------------------------------------------------------------------
// ab = bf16(ctx + h_seq)
// ---------------------------------------------------------------------------
__global__ void add_bf16_kernel(const float* __restrict__ a, const float* __restrict__ b,
                                bf16* __restrict__ o, size_t n) {
  size_t i = (size_t)blockIdx.x * blockDim.x + threadIdx.x;
  if (i < n) o[i] = (bf16)(a[i] + b[i]);
}

// ---------------------------------------------------------------------------
extern "C" void kernel_launch(void* const* d_in, const int* in_sizes, int n_in,
                              void* d_out, int out_size, void* d_ws, size_t ws_size,
                              hipStream_t stream) {
  (void)in_sizes; (void)n_in; (void)out_size; (void)ws_size;
  const int B = 2, S = 2048, D = 1024, H = 8, dh = 128, R = 256, F = 4096;
  const int BS = B * S;

  const float* x   = (const float*)d_in[0];
  const float* wn1 = (const float*)d_in[1];
  const float* wn2 = (const float*)d_in[2];
  const float* Wq  = (const float*)d_in[3];  const float* bq  = (const float*)d_in[4];
  const float* Wk  = (const float*)d_in[5];  const float* bk  = (const float*)d_in[6];
  const float* Wv  = (const float*)d_in[7];  const float* bv  = (const float*)d_in[8];
  const float* Wz  = (const float*)d_in[9];  const float* bz  = (const float*)d_in[10];
  const float* Wg  = (const float*)d_in[11]; const float* bg  = (const float*)d_in[12];
  const float* Wd  = (const float*)d_in[13]; const float* bd  = (const float*)d_in[14];
  const float* Wu  = (const float*)d_in[15]; const float* bu  = (const float*)d_in[16];
  const float* Wo  = (const float*)d_in[17]; const float* bo  = (const float*)d_in[18];
  const float* Wf1 = (const float*)d_in[19]; const float* bf1 = (const float*)d_in[20];
  const float* Wf2 = (const float*)d_in[21]; const float* bf2 = (const float*)d_in[22];

  // ---- workspace bump allocator (aliased where lifetimes allow) ----
  char* wsp = (char*)d_ws;
  auto alloc = [&](size_t bytes) -> char* {
    char* r = wsp;
    wsp += (bytes + 255) & ~(size_t)255;
    return r;
  };
  bf16* wqt  = (bf16*)alloc((size_t)D * D * 2);
  bf16* wkt  = (bf16*)alloc((size_t)D * D * 2);
  bf16* wvt  = (bf16*)alloc((size_t)D * D * 2);
  bf16* wzt  = (bf16*)alloc((size_t)D * D * 2);
  bf16* wgt  = (bf16*)alloc((size_t)D * D * 2);
  bf16* wot  = (bf16*)alloc((size_t)D * D * 2);
  bf16* wdt  = (bf16*)alloc((size_t)R * D * 2);   // [R][D]
  bf16* wut  = (bf16*)alloc((size_t)D * R * 2);   // [D][R]
  bf16* wf1t = (bf16*)alloc((size_t)F * D * 2);   // [F][D]
  bf16* wf2t = (bf16*)alloc((size_t)D * F * 2);   // [D][F]
  bf16* xn   = (bf16*)alloc((size_t)BS * D * 2);  // reused later as yn
  bf16* qb   = (bf16*)alloc((size_t)BS * D * 2);
  bf16* vb   = (bf16*)alloc((size_t)BS * D * 2);  // reused later as ab
  bf16* vt   = (bf16*)alloc((size_t)BS * D * 2);
  bf16* zb   = (bf16*)alloc((size_t)BS * D * 2);
  bf16* ksb  = (bf16*)alloc((size_t)BS * D * 2);
  bf16* zrb  = (bf16*)alloc((size_t)BS * R * 2);
  bf16* f1b  = (bf16*)alloc((size_t)BS * F * 2);
  float* kf  = (float*)alloc((size_t)BS * D * 4); // reused later as ctx
  float* zf  = (float*)alloc((size_t)BS * D * 4);
  float* gam = (float*)alloc((size_t)BS * D * 4); // reused later as y
  float* hs  = (float*)alloc((size_t)BS * D * 4);
  bf16* yn = xn;       // xn dead after z GEMM
  bf16* ab = vb;       // vb dead after v transpose
  float* ctx = kf;     // kf dead after scan
  float* y   = gam;    // gamma dead after recurrence/scan

  const size_t nBD = (size_t)BS * D;

  // 1) weight transposes (fp32 -> bf16 [N][K])
  auto tgrid = [](size_t n) { return (unsigned)((n + 255) / 256); };
  transpose_w_kernel<<<tgrid((size_t)D * D), 256, 0, stream>>>(Wq,  wqt,  D, D);
  transpose_w_kernel<<<tgrid((size_t)D * D), 256, 0, stream>>>(Wk,  wkt,  D, D);
  transpose_w_kernel<<<tgrid((size_t)D * D), 256, 0, stream>>>(Wv,  wvt,  D, D);
  transpose_w_kernel<<<tgrid((size_t)D * D), 256, 0, stream>>>(Wz,  wzt,  D, D);
  transpose_w_kernel<<<tgrid((size_t)D * D), 256, 0, stream>>>(Wg,  wgt,  D, D);
  transpose_w_kernel<<<tgrid((size_t)D * D), 256, 0, stream>>>(Wo,  wot,  D, D);
  transpose_w_kernel<<<tgrid((size_t)D * R), 256, 0, stream>>>(Wd,  wdt,  D, R);
  transpose_w_kernel<<<tgrid((size_t)R * D), 256, 0, stream>>>(Wu,  wut,  R, D);
  transpose_w_kernel<<<tgrid((size_t)D * F), 256, 0, stream>>>(Wf1, wf1t, D, F);
  transpose_w_kernel<<<tgrid((size_t)F * D), 256, 0, stream>>>(Wf2, wf2t, F, D);

  // 2) RMSNorm(x) -> bf16
  rmsnorm_kernel<<<BS, 256, 0, stream>>>(x, wn1, xn, D);

  // 3) Q/K/V/Z projections (WMMA, async-LDS staged B)
  dim3 gD(D / 64, BS / 64), gR(R / 64, BS / 64), gF(F / 64, BS / 64);
  gemm_wmma_kernel<EPI_NONE><<<gD, 128, 0, stream>>>(xn, wqt, bq, nullptr, nullptr, qb, BS, D, D);
  gemm_wmma_kernel<EPI_NONE><<<gD, 128, 0, stream>>>(xn, wkt, bk, nullptr, kf, nullptr, BS, D, D);
  gemm_wmma_kernel<EPI_NONE><<<gD, 128, 0, stream>>>(xn, wvt, bv, nullptr, nullptr, vb, BS, D, D);
  gemm_wmma_kernel<EPI_NONE><<<gD, 128, 0, stream>>>(xn, wzt, bz, nullptr, zf, zb, BS, D, D);

  // 4) V -> [B,H,dh,S] for PV fragments
  transpose_v_kernel<<<tgrid(nBD), 256, 0, stream>>>(vb, vt, B, S, D, dh);

  // 5) decay MLP: gamma = clip(sigmoid(relu(z@Wd+bd)@Wu+bu))
  gemm_wmma_kernel<EPI_RELU><<<gR, 128, 0, stream>>>(zb, wdt, bd, nullptr, nullptr, zrb, BS, R, D);
  gemm_wmma_kernel<EPI_SIGCLIP><<<gD, 128, 0, stream>>>(zrb, wut, bu, nullptr, gam, nullptr, BS, D, R);

  // 6) associative scan (log-cumsum) + k scaling
  scan_scale_kernel<<<B, D, 0, stream>>>(gam, kf, ksb, S, D);

  // 7) gated recurrence h_t
  recurrence_kernel<<<B, D, 0, stream>>>(zf, gam, wgt, bg, hs, S, D);

  // 8) causal flash attention (WMMA)
  attn_kernel<<<dim3(S / 16, H, B), 32, 0, stream>>>(qb, ksb, vt, ctx, B, H, S, dh);

  // 9) y = x + (ctx + h) @ Wo + bo
  add_bf16_kernel<<<tgrid(nBD), 256, 0, stream>>>(ctx, hs, ab, nBD);
  gemm_wmma_kernel<EPI_RESID><<<gD, 128, 0, stream>>>(ab, wot, bo, x, y, nullptr, BS, D, D);

  // 10) FFN with pre-norm + residual
  rmsnorm_kernel<<<BS, 256, 0, stream>>>(y, wn2, yn, D);
  gemm_wmma_kernel<EPI_GELU><<<gF, 128, 0, stream>>>(yn, wf1t, bf1, nullptr, nullptr, f1b, BS, F, D);
  gemm_wmma_kernel<EPI_RESID><<<gD, 128, 0, stream>>>(f1b, wf2t, bf2, y, (float*)d_out, nullptr, BS, D, F);
}
